// SwinBlock_40475771797518
// MI455X (gfx1250) — compile-verified, matrix operand order checked by reading, not run on hardware
//
#include <hip/hip_runtime.h>
#include <math.h>

// ---------------------------------------------------------------------------
// Types for CDNA5 WMMA (wave32, 16x16x32 bf16 -> f32)
// ---------------------------------------------------------------------------
typedef __attribute__((ext_vector_type(16))) __bf16 v16bf;
typedef __attribute__((ext_vector_type(8)))  __bf16 v8bf;
typedef __attribute__((ext_vector_type(8)))  float  v8f;

// Fragment loader for BOTH A (16x32, M-major rows) and B^T (16x32, N-major rows)
// stored row-major [16+][stride] bf16 in LDS.
// Per ISA 16-bit A layout: lanes 0-15 hold K=0..7 (VGPR0-3) and K=16..23 (VGPR4-7);
// lanes 16-31 hold K=8..15 and K=24..31.  Two b128 LDS loads per lane.
__device__ __forceinline__ v16bf load_frag(const __bf16* base, int stride,
                                           int rowBase, int kOff) {
    int lane = threadIdx.x & 31;
    const __bf16* p = base + (size_t)(rowBase + (lane & 15)) * stride
                           + kOff + ((lane >> 4) << 3);
    v8bf lo = *(const v8bf*)p;
    v8bf hi = *(const v8bf*)(p + 16);
    return __builtin_shufflevector(lo, hi, 0,1,2,3,4,5,6,7,8,9,10,11,12,13,14,15);
}

__device__ __forceinline__ v8f wmma_bf16(v16bf a, v16bf b, v8f c) {
    return __builtin_amdgcn_wmma_f32_16x16x32_bf16(
        false, a, false, b, (short)0, c, false, false);
}

// CDNA5 async global->LDS DMA (per-lane b128), tracked by ASYNCcnt.
// VDST holds the wave-relative LDS byte offset = low 32 bits of the generic
// shared-space address.
__device__ __forceinline__ void async_b128(const void* g, void* l) {
    unsigned loff = (unsigned)(unsigned long long)l;
    asm volatile("global_load_async_to_lds_b128 %0, %1, off"
                 :: "v"(loff), "v"(g) : "memory");
}
__device__ __forceinline__ void wait_async0() {
    asm volatile("s_wait_asynccnt 0" ::: "memory");
}

// ---------------------------------------------------------------------------
// Constants for this problem
// ---------------------------------------------------------------------------
#define CB  512          // channels
#define HH  96           // height/width
#define WSZ 6            // window size
#define NTOK 36          // tokens per window
#define NWIN 4096        // B * 16 * 16 windows
#define MTOT 147456      // NWIN * NTOK tokens
#define HEADS 16
#define HD  32           // head dim
#define LSTR 72          // LDS row stride (bf16) for 64-wide k-slabs

// ---------------------------------------------------------------------------
// Weight transpose + fp32 -> bf16 cast:   W[K][N]  ->  Wt[N][K]
// ---------------------------------------------------------------------------
__global__ void transpose_cast_kernel(const float* __restrict__ W,
                                      __bf16* __restrict__ Wt, int K, int N) {
    int idx = blockIdx.x * 256 + threadIdx.x;
    if (idx >= K * N) return;
    int n = idx / K, k = idx - n * K;
    Wt[idx] = (__bf16)W[(size_t)k * N + n];
}

// ---------------------------------------------------------------------------
// LayerNorm (one wave32 per 512-channel token), optional window-gather, ->bf16
// ---------------------------------------------------------------------------
template<int GATHER>
__global__ __launch_bounds__(256) void ln_kernel(const float* __restrict__ src,
                                                 const float* __restrict__ g,
                                                 const float* __restrict__ b,
                                                 __bf16* __restrict__ out) {
    int lane = threadIdx.x & 31;
    int m = blockIdx.x * 8 + (threadIdx.x >> 5);
    const float* row;
    if (GATHER) {
        int win = m / NTOK, n = m - win * NTOK;
        int wc = win & 15, wh = (win >> 4) & 15, bb = win >> 8;
        int r = n / WSZ, c = n - r * WSZ;
        int hh = wh * WSZ + r, ww = wc * WSZ + c;
        row = src + (((size_t)bb * HH + hh) * HH + ww) * CB;
    } else {
        row = src + (size_t)m * CB;
    }
    float vals[16];
    float s = 0.f, sq = 0.f;
    #pragma unroll
    for (int i = 0; i < 16; ++i) {
        float v = row[lane + 32 * i];
        vals[i] = v; s += v; sq += v * v;
    }
    #pragma unroll
    for (int off = 16; off >= 1; off >>= 1) {
        s  += __shfl_xor(s, off, 32);
        sq += __shfl_xor(sq, off, 32);
    }
    float mean = s * (1.f / CB);
    float var  = sq * (1.f / CB) - mean * mean;
    float rstd = rsqrtf(var + 1e-5f);
    __bf16* orow = out + (size_t)m * CB;
    #pragma unroll
    for (int i = 0; i < 16; ++i) {
        int col = lane + 32 * i;
        orow[col] = (__bf16)((vals[i] - mean) * rstd * g[col] + b[col]);
    }
}

// ---------------------------------------------------------------------------
// Tiled WMMA GEMM:  C[M][N] = A[M][K](bf16) * Bt[N][K](bf16)^T  + epilogue
//   block = 256 threads = 8 waves; tile 128x128; k-slab = 64;
//   double-buffered LDS filled by async global->LDS DMA (ASYNCcnt).
//   8 waves as 4(M) x 2(N); each wave: 2x4 16x16 f32 accum tiles.
// EPI: 0 = store bf16 (qkv)
//      1 = +bias, store f32 (proj -> x2)
//      2 = +bias, exact GELU, store bf16 (fc1 -> h1)
//      3 = +bias, +resid(window layout), un-window permute, store f32 (fc2 -> out)
// ---------------------------------------------------------------------------
template<int EPI>
__global__ __launch_bounds__(256) void gemm_kernel(const __bf16* __restrict__ A,
                                                   const __bf16* __restrict__ Bt,
                                                   const float* __restrict__ bias,
                                                   const float* __restrict__ resid,
                                                   void* __restrict__ Cout,
                                                   int M, int N, int K) {
    __shared__ __bf16 As[2][128 * LSTR];   // 18 KB each buffer
    __shared__ __bf16 Bs[2][128 * LSTR];
    int tid  = threadIdx.x;
    int lane = tid & 31, wid = tid >> 5;
    int wm = wid & 3, wn = wid >> 2;
    size_t rowBase = (size_t)blockIdx.y * 128;
    int    colBase = blockIdx.x * 128;

    v8f acc[2][4];
    #pragma unroll
    for (int i = 0; i < 2; ++i)
        #pragma unroll
        for (int j = 0; j < 4; ++j) acc[i][j] = (v8f)0.f;

    // stage one 128x64 slab of A and Bt into LDS buffer `buf` via async DMA
    auto stage = [&](int buf, int k0) {
        #pragma unroll
        for (int s = 0; s < 4; ++s) {
            int chunk = tid + s * 256;            // 1024 chunks of 8 bf16
            int row = chunk >> 3, q = (chunk & 7) * 8;
            async_b128(&A[(rowBase + row) * (size_t)K + k0 + q],
                       &As[buf][row * LSTR + q]);
            async_b128(&Bt[((size_t)colBase + row) * (size_t)K + k0 + q],
                       &Bs[buf][row * LSTR + q]);
        }
    };

    stage(0, 0);
    wait_async0();
    __syncthreads();
    int cur = 0;
    for (int k0 = 0; k0 < K; k0 += 64) {
        if (k0 + 64 < K) stage(cur ^ 1, k0 + 64);   // prefetch next slab
        #pragma unroll
        for (int ks = 0; ks < 64; ks += 32) {
            v16bf af[2], bf[4];
            #pragma unroll
            for (int i = 0; i < 2; ++i)
                af[i] = load_frag(&As[cur][0], LSTR, wm * 32 + i * 16, ks);
            #pragma unroll
            for (int j = 0; j < 4; ++j)
                bf[j] = load_frag(&Bs[cur][0], LSTR, wn * 64 + j * 16, ks);
            #pragma unroll
            for (int i = 0; i < 2; ++i)
                #pragma unroll
                for (int j = 0; j < 4; ++j)
                    acc[i][j] = wmma_bf16(af[i], bf[j], acc[i][j]);
        }
        wait_async0();       // next slab landed in LDS
        __syncthreads();     // all waves done reading cur, next visible to all
        cur ^= 1;
    }

    // Epilogue.  C/D layout: VGPR jr, lanes 0-15: M=jr, N=lane; lanes 16-31: M=8+jr.
    int nloc = lane & 15, mbase = (lane >> 4) * 8;
    #pragma unroll
    for (int i = 0; i < 2; ++i) {
        #pragma unroll
        for (int j = 0; j < 4; ++j) {
            int gn = colBase + wn * 64 + j * 16 + nloc;
            float bv = (EPI != 0) ? bias[gn] : 0.f;
            #pragma unroll
            for (int jr = 0; jr < 8; ++jr) {
                size_t gm = rowBase + wm * 32 + i * 16 + mbase + jr;
                float v = acc[i][j][jr] + bv;
                if (EPI == 0) {
                    ((__bf16*)Cout)[gm * (size_t)N + gn] = (__bf16)v;
                } else if (EPI == 1) {
                    ((float*)Cout)[gm * (size_t)N + gn] = v;
                } else if (EPI == 2) {
                    v = 0.5f * v * (1.f + erff(v * 0.70710678118654752f));
                    ((__bf16*)Cout)[gm * (size_t)N + gn] = (__bf16)v;
                } else {
                    v += resid[gm * (size_t)CB + gn];
                    int win = (int)(gm / NTOK), n = (int)(gm - (size_t)win * NTOK);
                    int wc = win & 15, wh = (win >> 4) & 15, bb = win >> 8;
                    int r = n / WSZ, c = n - r * WSZ;
                    size_t oidx = (((size_t)bb * HH + wh * WSZ + r) * HH
                                   + wc * WSZ + c) * CB + gn;
                    ((float*)Cout)[oidx] = v;
                }
            }
        }
    }
}

// ---------------------------------------------------------------------------
// Windowed attention: one wave32 per (window, head).
//   S = (q k^T) * d^-0.5 + rel_bias ; P = softmax(S) ; O = P v
//   36 tokens padded to 48 (M/N) and 64 (K of PV); pads zeroed.
// ---------------------------------------------------------------------------
__global__ __launch_bounds__(32) void attn_kernel(const __bf16* __restrict__ qkv,
                                                  const float* __restrict__ rel_bias,
                                                  __bf16* __restrict__ o) {
    __shared__ __bf16 qs[48 * 32];
    __shared__ __bf16 ks[48 * 32];
    __shared__ __bf16 vt[32 * 64];   // V^T: [dim][token], token padded to 64
    __shared__ float  Ss[48 * 49];
    __shared__ __bf16 Ps[48 * 64];   // P: [row][token], K padded to 64

    int lane = threadIdx.x;
    int win = blockIdx.x, head = blockIdx.y;
    size_t rowb = (size_t)win * NTOK * (3 * CB);
    int qc = head * HD, kc = CB + head * HD, vc = 2 * CB + head * HD;

    __bf16 z = (__bf16)0.f;
    for (int i = NTOK * 32 + lane; i < 48 * 32; i += 32) { qs[i] = z; ks[i] = z; }
    for (int t = NTOK; t < 64; ++t) vt[lane * 64 + t] = z;
    for (int i = lane; i < 48 * 64; i += 32) Ps[i] = z;

    for (int c = lane; c < NTOK * 4; c += 32) {       // 4 x 8 bf16 chunks per row
        int row = c >> 2, q8 = (c & 3) * 8;
        const __bf16* src = qkv + rowb + (size_t)row * (3 * CB);
        *(uint4*)&qs[row * 32 + q8] = *(const uint4*)&src[qc + q8];
        *(uint4*)&ks[row * 32 + q8] = *(const uint4*)&src[kc + q8];
    }
    for (int t = 0; t < NTOK; ++t)                    // lane = head dim
        vt[lane * 64 + t] = qkv[rowb + (size_t)t * (3 * CB) + vc + lane];
    __syncthreads();

    // ---- S = q k^T : 3x3 tiles, K = 32 = head dim (single WMMA step) ----
    v16bf aq[3], bk[3];
    #pragma unroll
    for (int i = 0; i < 3; ++i) { aq[i] = load_frag(qs, 32, i * 16, 0);
                                  bk[i] = load_frag(ks, 32, i * 16, 0); }
    int nloc = lane & 15, mbase = (lane >> 4) * 8;
    #pragma unroll
    for (int mi = 0; mi < 3; ++mi)
        #pragma unroll
        for (int ni = 0; ni < 3; ++ni) {
            v8f sacc = (v8f)0.f;
            sacc = wmma_bf16(aq[mi], bk[ni], sacc);
            #pragma unroll
            for (int jr = 0; jr < 8; ++jr)
                Ss[(mi * 16 + mbase + jr) * 49 + ni * 16 + nloc] = sacc[jr];
        }
    __syncthreads();

    // ---- softmax with relative position bias (rows 0..35) ----
    const float scale = 0.17677669529663687f;   // 32^-0.5
    for (int r = lane; r < NTOK; r += 32) {
        int ri = r / WSZ, ci = r - ri * WSZ;
        float mx = -1e30f;
        for (int c = 0; c < NTOK; ++c) {
            int rj = c / WSZ, cj = c - rj * WSZ;
            int bidx = (ri - rj + 5) * 11 + (ci - cj + 5);
            float v = Ss[r * 49 + c] * scale + rel_bias[bidx * HEADS + head];
            Ss[r * 49 + c] = v;
            mx = fmaxf(mx, v);
        }
        float sum = 0.f;
        for (int c = 0; c < NTOK; ++c) {
            float e = expf(Ss[r * 49 + c] - mx);
            Ss[r * 49 + c] = e;
            sum += e;
        }
        float inv = 1.f / sum;
        for (int c = 0; c < NTOK; ++c)
            Ps[r * 64 + c] = (__bf16)(Ss[r * 49 + c] * inv);
    }
    __syncthreads();

    // ---- O = P @ V : 3 M-tiles x 2 N-tiles, K = 64 (2 WMMA steps) ----
    #pragma unroll
    for (int mi = 0; mi < 3; ++mi) {
        v16bf ap0 = load_frag(Ps, 64, mi * 16, 0);
        v16bf ap1 = load_frag(Ps, 64, mi * 16, 32);
        #pragma unroll
        for (int nj = 0; nj < 2; ++nj) {
            v8f oacc = (v8f)0.f;
            oacc = wmma_bf16(ap0, load_frag(vt, 64, nj * 16, 0),  oacc);
            oacc = wmma_bf16(ap1, load_frag(vt, 64, nj * 16, 32), oacc);
            #pragma unroll
            for (int jr = 0; jr < 8; ++jr) {
                int m = mi * 16 + mbase + jr;
                if (m < NTOK)
                    o[((size_t)win * NTOK + m) * CB + head * HD + nj * 16 + nloc]
                        = (__bf16)oacc[jr];
            }
        }
    }
}

// ---------------------------------------------------------------------------
// Launcher
// ---------------------------------------------------------------------------
extern "C" void kernel_launch(void* const* d_in, const int* in_sizes, int n_in,
                              void* d_out, int out_size, void* d_ws, size_t ws_size,
                              hipStream_t stream) {
    const float* x       = (const float*)d_in[0];
    const float* w_qkv   = (const float*)d_in[1];
    const float* w_proj  = (const float*)d_in[2];
    const float* b_proj  = (const float*)d_in[3];
    const float* relbias = (const float*)d_in[4];
    const float* ln1_g   = (const float*)d_in[5];
    const float* ln1_b   = (const float*)d_in[6];
    const float* ln2_g   = (const float*)d_in[7];
    const float* ln2_b   = (const float*)d_in[8];
    const float* w_fc1   = (const float*)d_in[9];
    const float* b_fc1   = (const float*)d_in[10];
    const float* w_fc2   = (const float*)d_in[11];
    const float* b_fc2   = (const float*)d_in[12];
    float* out = (float*)d_out;

    char* ws = (char*)d_ws;
    size_t off = 0;
    auto alloc = [&](size_t bytes) {
        size_t o = off; off += (bytes + 255) & ~(size_t)255; return o;
    };
    __bf16* wqkvT  = (__bf16*)(ws + alloc((size_t)1536 * 512 * 2));
    __bf16* wprojT = (__bf16*)(ws + alloc((size_t)512 * 512 * 2));
    __bf16* wfc1T  = (__bf16*)(ws + alloc((size_t)2048 * 512 * 2));
    __bf16* wfc2T  = (__bf16*)(ws + alloc((size_t)512 * 2048 * 2));
    __bf16* xn     = (__bf16*)(ws + alloc((size_t)MTOT * 512 * 2));   // reused as xm
    __bf16* qkv    = (__bf16*)(ws + alloc((size_t)MTOT * 2048 * 2));  // reused as h1
    __bf16* obuf   = (__bf16*)(ws + alloc((size_t)MTOT * 512 * 2));
    float*  x2     = (float*) (ws + alloc((size_t)MTOT * 512 * 4));
    __bf16* xm = xn;
    __bf16* h1 = qkv;

    // 0) weight transpose + cast
    transpose_cast_kernel<<<(512 * 1536 + 255) / 256, 256, 0, stream>>>(w_qkv, wqkvT, 512, 1536);
    transpose_cast_kernel<<<(512 * 512  + 255) / 256, 256, 0, stream>>>(w_proj, wprojT, 512, 512);
    transpose_cast_kernel<<<(512 * 2048 + 255) / 256, 256, 0, stream>>>(w_fc1, wfc1T, 512, 2048);
    transpose_cast_kernel<<<(2048 * 512 + 255) / 256, 256, 0, stream>>>(w_fc2, wfc2T, 2048, 512);

    // 1) LN1 + window partition (gather) -> xn bf16, window-ordered
    ln_kernel<1><<<MTOT / 8, 256, 0, stream>>>(x, ln1_g, ln1_b, xn);

    // 2) qkv = xn @ w_qkv              (M x 512 x 1536, bf16 out)
    gemm_kernel<0><<<dim3(1536 / 128, MTOT / 128), 256, 0, stream>>>(
        xn, wqkvT, nullptr, nullptr, qkv, MTOT, 1536, 512);

    // 3) windowed attention -> obuf bf16
    attn_kernel<<<dim3(NWIN, HEADS), 32, 0, stream>>>(qkv, relbias, obuf);

    // 4) x2 = obuf @ w_proj + b_proj   (f32, window layout)
    gemm_kernel<1><<<dim3(512 / 128, MTOT / 128), 256, 0, stream>>>(
        obuf, wprojT, b_proj, nullptr, x2, MTOT, 512, 512);

    // 5) LN2 -> xm bf16
    ln_kernel<0><<<MTOT / 8, 256, 0, stream>>>(x2, ln2_g, ln2_b, xm);

    // 6) h1 = gelu(xm @ w_fc1 + b_fc1) (bf16)
    gemm_kernel<2><<<dim3(2048 / 128, MTOT / 128), 256, 0, stream>>>(
        xm, wfc1T, b_fc1, nullptr, h1, MTOT, 2048, 512);

    // 7) out = x2 + h1 @ w_fc2 + b_fc2, un-windowed to (B,H,W,C)
    gemm_kernel<3><<<dim3(512 / 128, MTOT / 128), 256, 0, stream>>>(
        h1, wfc2T, b_fc2, x2, out, MTOT, 512, 2048);
}
